// MessagePassingLayer_13915694039743
// MI455X (gfx1250) — compile-verified, matrix operand order checked by reading, not run on hardware
//
#include <hip/hip_runtime.h>
#include <hip/hip_bf16.h>

#define NN 20000
#define NE 320000
#define ND 256
#define ED 64
#define HID 512
#define EDGE_IN (2*ND + ED)   // 576
#define NODE_IN (ND + HID)    // 768

typedef __attribute__((ext_vector_type(16))) __bf16 v16bf;
typedef __attribute__((ext_vector_type(8)))  float  v8f;

union FragBF { v16bf v; uint4 q[2]; };

__device__ __forceinline__ unsigned short f2bf(float f) {
  union { float f; unsigned int u; } cv; cv.f = f;
  unsigned int u = cv.u;
  u += 0x7FFFu + ((u >> 16) & 1u);          // round-to-nearest-even
  return (unsigned short)(u >> 16);
}

// Load a 16x32 bf16 A- (or Kx16 B-) fragment: two 16B chunks per lane.
// p must already include: row*stride + kbase + ((lane&16)?8:0)
__device__ __forceinline__ v16bf load_frag(const unsigned short* p) {
  FragBF f;
  f.q[0] = *(const uint4*)(p);
  f.q[1] = *(const uint4*)(p + 16);
  return f.v;
}

__device__ __forceinline__ v8f wmma_bf16(v16bf a, v16bf b, v8f c) {
  return __builtin_amdgcn_wmma_f32_16x16x32_bf16(false, a, false, b, (short)0, c,
                                                 false, false);
}

// dst[n*K + k] = bf16(src[k*N + n])  (store W transposed, bf16)
__global__ void transpose_bf16_kernel(const float* __restrict__ src,
                                      unsigned short* __restrict__ dst,
                                      int K, int N) {
  int total = K * N;
  for (int i = blockIdx.x * blockDim.x + threadIdx.x; i < total;
       i += gridDim.x * blockDim.x) {
    int n = i / K;
    int k = i - n * K;
    dst[i] = f2bf(src[(size_t)k * N + n]);
  }
}

__global__ void zero_kernel(float4* __restrict__ p, int n4) {
  float4 z = make_float4(0.f, 0.f, 0.f, 0.f);
  for (int i = blockIdx.x * blockDim.x + threadIdx.x; i < n4;
       i += gridDim.x * blockDim.x)
    p[i] = z;
}

// ---------------- edge kernel: 64 edges per block, 8 waves --------------
__global__ void __launch_bounds__(256)
edge_kernel(const float* __restrict__ nodef, const int* __restrict__ eidx,
            const float* __restrict__ edgef,
            const unsigned short* __restrict__ W1t, const float* __restrict__ b1,
            const unsigned short* __restrict__ W2t, const float* __restrict__ b2,
            float* __restrict__ agg) {
  __shared__ __align__(16) unsigned short Xs[64 * EDGE_IN];   // 73728 B
  __shared__ __align__(16) unsigned short Hs[64 * HID];       // 65536 B
  __shared__ int sIdxS[64];
  __shared__ int tIdxS[64];

  const int tid  = threadIdx.x;
  const int base = blockIdx.x * 64;

  if (tid < 64) {
    sIdxS[tid] = eidx[base + tid];
    tIdxS[tid] = eidx[NE + base + tid];
  }
  __syncthreads();

  // gather endpoints + edge feats, convert to bf16 in LDS
  for (int i = tid; i < 64 * (EDGE_IN / 4); i += 256) {
    int row = i / (EDGE_IN / 4);
    int col = (i - row * (EDGE_IN / 4)) * 4;
    const float* sp;
    if (col < ND)          sp = nodef + (size_t)sIdxS[row] * ND + col;
    else if (col < 2 * ND) sp = nodef + (size_t)tIdxS[row] * ND + (col - ND);
    else                   sp = edgef + (size_t)(base + row) * ED + (col - 2 * ND);
    float4 v = *(const float4*)sp;
    ushort4 h;
    h.x = f2bf(v.x); h.y = f2bf(v.y); h.z = f2bf(v.z); h.w = f2bf(v.w);
    *(ushort4*)&Xs[row * EDGE_IN + col] = h;
  }
  __syncthreads();

  const int w     = tid >> 5;
  const int lane  = tid & 31;
  const int lrow  = lane & 15;
  const int khalf = (lane & 16) ? 8 : 0;
  const int mrow  = (lane & 16) ? 8 : 0;

  // ---- H = relu(X @ W1 + b1) ----
  v8f acc[4][4] = {};
  for (int k = 0; k < EDGE_IN; k += 32) {
    v16bf Afr[4], Bfr[4];
#pragma unroll
    for (int mt = 0; mt < 4; ++mt)
      Afr[mt] = load_frag(&Xs[(mt * 16 + lrow) * EDGE_IN + k + khalf]);
#pragma unroll
    for (int j = 0; j < 4; ++j) {
      int n = (w * 4 + j) * 16 + lrow;
      Bfr[j] = load_frag(&W1t[(size_t)n * EDGE_IN + k + khalf]);
    }
#pragma unroll
    for (int mt = 0; mt < 4; ++mt)
#pragma unroll
      for (int j = 0; j < 4; ++j)
        acc[mt][j] = wmma_bf16(Afr[mt], Bfr[j], acc[mt][j]);
  }
#pragma unroll
  for (int j = 0; j < 4; ++j) {
    int n = (w * 4 + j) * 16 + lrow;
    float bias = b1[n];
#pragma unroll
    for (int mt = 0; mt < 4; ++mt)
#pragma unroll
      for (int r = 0; r < 8; ++r) {
        float x = acc[mt][j][r] + bias;
        x = x > 0.f ? x : 0.f;
        Hs[(mt * 16 + mrow + r) * HID + n] = f2bf(x);
      }
  }
  __syncthreads();

  // ---- M = H @ W2 + b2, scatter-add to agg[tgt] ----
  v8f acc2[4][4] = {};
  for (int k = 0; k < HID; k += 32) {
    v16bf Afr[4], Bfr[4];
#pragma unroll
    for (int mt = 0; mt < 4; ++mt)
      Afr[mt] = load_frag(&Hs[(mt * 16 + lrow) * HID + k + khalf]);
#pragma unroll
    for (int j = 0; j < 4; ++j) {
      int n = (w * 4 + j) * 16 + lrow;
      Bfr[j] = load_frag(&W2t[(size_t)n * HID + k + khalf]);
    }
#pragma unroll
    for (int mt = 0; mt < 4; ++mt)
#pragma unroll
      for (int j = 0; j < 4; ++j)
        acc2[mt][j] = wmma_bf16(Afr[mt], Bfr[j], acc2[mt][j]);
  }
#pragma unroll
  for (int j = 0; j < 4; ++j) {
    int n = (w * 4 + j) * 16 + lrow;
    float bias = b2[n];
#pragma unroll
    for (int mt = 0; mt < 4; ++mt)
#pragma unroll
      for (int r = 0; r < 8; ++r) {
        int m = mt * 16 + mrow + r;
        float val = acc2[mt][j][r] + bias;
        int t = tIdxS[m];
        __hip_atomic_fetch_add(&agg[(size_t)t * HID + n], val,
                               __ATOMIC_RELAXED, __HIP_MEMORY_SCOPE_AGENT);
      }
  }
}

// ---------------- node kernel: 64 nodes per block, 8 waves --------------
__global__ void __launch_bounds__(256)
node_kernel(const float* __restrict__ nodef, const float* __restrict__ agg,
            const unsigned short* __restrict__ W3t, const float* __restrict__ b3,
            const unsigned short* __restrict__ W4t, const float* __restrict__ b4,
            float* __restrict__ out) {
  __shared__ __align__(16) unsigned short Xs[64 * NODE_IN];   // 98304 B
  __shared__ __align__(16) unsigned short Gs[64 * HID];       // 65536 B

  const int tid  = threadIdx.x;
  const int base = blockIdx.x * 64;

  for (int i = tid; i < 64 * (NODE_IN / 4); i += 256) {
    int row  = i / (NODE_IN / 4);
    int col  = (i - row * (NODE_IN / 4)) * 4;
    int node = base + row;
    float4 v = make_float4(0.f, 0.f, 0.f, 0.f);
    if (node < NN) {
      const float* sp = (col < ND) ? (nodef + (size_t)node * ND + col)
                                   : (agg + (size_t)node * HID + (col - ND));
      v = *(const float4*)sp;
    }
    ushort4 h;
    h.x = f2bf(v.x); h.y = f2bf(v.y); h.z = f2bf(v.z); h.w = f2bf(v.w);
    *(ushort4*)&Xs[row * NODE_IN + col] = h;
  }
  __syncthreads();

  const int w     = tid >> 5;
  const int lane  = tid & 31;
  const int lrow  = lane & 15;
  const int khalf = (lane & 16) ? 8 : 0;
  const int mrow  = (lane & 16) ? 8 : 0;

  // ---- G = relu(X2 @ W3 + b3) ----
  v8f acc[4][4] = {};
  for (int k = 0; k < NODE_IN; k += 32) {
    v16bf Afr[4], Bfr[4];
#pragma unroll
    for (int mt = 0; mt < 4; ++mt)
      Afr[mt] = load_frag(&Xs[(mt * 16 + lrow) * NODE_IN + k + khalf]);
#pragma unroll
    for (int j = 0; j < 4; ++j) {
      int n = (w * 4 + j) * 16 + lrow;
      Bfr[j] = load_frag(&W3t[(size_t)n * NODE_IN + k + khalf]);
    }
#pragma unroll
    for (int mt = 0; mt < 4; ++mt)
#pragma unroll
      for (int j = 0; j < 4; ++j)
        acc[mt][j] = wmma_bf16(Afr[mt], Bfr[j], acc[mt][j]);
  }
#pragma unroll
  for (int j = 0; j < 4; ++j) {
    int n = (w * 4 + j) * 16 + lrow;
    float bias = b3[n];
#pragma unroll
    for (int mt = 0; mt < 4; ++mt)
#pragma unroll
      for (int r = 0; r < 8; ++r) {
        float x = acc[mt][j][r] + bias;
        x = x > 0.f ? x : 0.f;
        Gs[(mt * 16 + mrow + r) * HID + n] = f2bf(x);
      }
  }
  __syncthreads();

  // ---- U = G @ W4 + b4 ; out = node_features + U ----
  v8f acc2[4][2] = {};
  for (int k = 0; k < HID; k += 32) {
    v16bf Afr[4], Bfr[2];
#pragma unroll
    for (int mt = 0; mt < 4; ++mt)
      Afr[mt] = load_frag(&Gs[(mt * 16 + lrow) * HID + k + khalf]);
#pragma unroll
    for (int j = 0; j < 2; ++j) {
      int n = (w * 2 + j) * 16 + lrow;
      Bfr[j] = load_frag(&W4t[(size_t)n * HID + k + khalf]);
    }
#pragma unroll
    for (int mt = 0; mt < 4; ++mt)
#pragma unroll
      for (int j = 0; j < 2; ++j)
        acc2[mt][j] = wmma_bf16(Afr[mt], Bfr[j], acc2[mt][j]);
  }
#pragma unroll
  for (int j = 0; j < 2; ++j) {
    int n = (w * 2 + j) * 16 + lrow;
    float bias = b4[n];
#pragma unroll
    for (int mt = 0; mt < 4; ++mt)
#pragma unroll
      for (int r = 0; r < 8; ++r) {
        int m = mt * 16 + mrow + r;
        int node = base + m;
        if (node < NN) {
          size_t o = (size_t)node * ND + n;
          out[o] = nodef[o] + acc2[mt][j][r] + bias;
        }
      }
  }
}

extern "C" void kernel_launch(void* const* d_in, const int* in_sizes, int n_in,
                              void* d_out, int out_size, void* d_ws, size_t ws_size,
                              hipStream_t stream) {
  const float* nodef = (const float*)d_in[0];
  const int*   eidx  = (const int*)d_in[1];
  const float* edgef = (const float*)d_in[2];
  const float* W1 = (const float*)d_in[3];
  const float* b1 = (const float*)d_in[4];
  const float* W2 = (const float*)d_in[5];
  const float* b2 = (const float*)d_in[6];
  const float* W3 = (const float*)d_in[7];
  const float* b3 = (const float*)d_in[8];
  const float* W4 = (const float*)d_in[9];
  const float* b4 = (const float*)d_in[10];
  float* out = (float*)d_out;

  // workspace layout: bf16-transposed weights, then f32 aggregate buffer
  unsigned short* W1t = (unsigned short*)d_ws;           // [512][576]
  unsigned short* W2t = W1t + (size_t)HID * EDGE_IN;     // [512][512]
  unsigned short* W3t = W2t + (size_t)HID * HID;         // [512][768]
  unsigned short* W4t = W3t + (size_t)HID * NODE_IN;     // [256][512]
  float* agg = (float*)(W4t + (size_t)ND * HID);         // [20000][512]

  transpose_bf16_kernel<<<256, 256, 0, stream>>>(W1, W1t, EDGE_IN, HID);
  transpose_bf16_kernel<<<256, 256, 0, stream>>>(W2, W2t, HID, HID);
  transpose_bf16_kernel<<<256, 256, 0, stream>>>(W3, W3t, NODE_IN, HID);
  transpose_bf16_kernel<<<256, 256, 0, stream>>>(W4, W4t, HID, ND);
  zero_kernel<<<1024, 256, 0, stream>>>((float4*)agg, NN * HID / 4);
  edge_kernel<<<NE / 64, 256, 0, stream>>>(nodef, eidx, edgef, W1t, b1, W2t, b2, agg);
  node_kernel<<<(NN + 63) / 64, 256, 0, stream>>>(nodef, agg, W3t, b3, W4t, b4, out);
}
